// RetNetAttention_36472862277842
// MI455X (gfx1250) — matrix operand order repaired and checked by reading, compiled, not run on hardware
//
#include <hip/hip_runtime.h>
#include <hip/hip_bf16.h>
#include <math.h>

// ---------------------------------------------------------------------------
// RetNet attention for MI455X (gfx1250, wave32, WMMA).
// Pipeline: f32->bf16 converts, QKV GEMM (bf16 WMMA, f32 accum, async
// global->LDS DMA double-buffered), chunkwise retention (C=64, all inner
// products as 16x16x32 bf16 WMMA), output GEMM.
// ---------------------------------------------------------------------------

typedef __attribute__((ext_vector_type(16))) __bf16 bf16x16;
typedef __attribute__((ext_vector_type(8)))  __bf16 bf16x8;
typedef __attribute__((ext_vector_type(8)))  float  floatx8;

#define SCALE_QK 0.125f   // 1/sqrt(64)
#define LDT 72            // retention LDS row stride in halves (144B)
#define LDG 40            // gemm LDS row stride in halves (80B, 16B aligned)

__device__ __forceinline__ floatx8 wmma_bf16(bf16x16 a, bf16x16 b, floatx8 c) {
  return __builtin_amdgcn_wmma_f32_16x16x32_bf16(
      /*neg_a=*/false, a, /*neg_b=*/false, b,
      /*c_mod=*/(short)0, c, /*reuse_a=*/false, /*reuse_b=*/false);
}

// A fragment (M x K = 16 x 32, bf16) from row-major LDS tile (rows=M).
// lanes 0-15: M=lane,   K = k0+{0..7, 16..23}
// lanes16-31: M=lane-16,K = k0+{8..15,24..31}
__device__ __forceinline__ bf16x16 frag_a(const __bf16* base, int stride,
                                          int m0, int k0, int lane) {
  int m  = m0 + (lane & 15);
  int kb = k0 + ((lane >> 4) << 3);
  const __bf16* p = base + m * stride + kb;
  union { bf16x16 v; bf16x8 h[2]; } u;
  u.h[0] = *(const bf16x8*)(p);
  u.h[1] = *(const bf16x8*)(p + 16);
  return u.v;
}

// B fragment (K x N = 32 x 16, bf16) from N-major LDS tile (rows=N, cols=K).
// lanes 0-15: N=lane,   K = k0+{0..7, 16..23} ; lanes16-31: N=lane-16, K = k0+{8..15,24..31}
__device__ __forceinline__ bf16x16 frag_b(const __bf16* base, int stride,
                                          int n0, int k0, int lane) {
  int n  = n0 + (lane & 15);
  int kb = k0 + ((lane >> 4) << 3);
  const __bf16* p = base + n * stride + kb;
  union { bf16x16 v; bf16x8 h[2]; } u;
  u.h[0] = *(const bf16x8*)(p);
  u.h[1] = *(const bf16x8*)(p + 16);
  return u.v;
}

// ---- CDNA5 async global->LDS DMA (ASYNCcnt-tracked, no VGPR staging) ------
__device__ __forceinline__ uint32_t lds_addr32(const void* p) {
  // generic LDS pointer: low 32 bits are the wave-relative LDS byte offset
  return (uint32_t)(uintptr_t)p;
}

__device__ __forceinline__ void async_copy_b128(uint32_t lds_byte, const void* g) {
  unsigned long long ga = (unsigned long long)(uintptr_t)g;
  asm volatile("global_load_async_to_lds_b128 %0, %1, off"
               :: "v"(lds_byte), "v"(ga) : "memory");
}

__device__ __forceinline__ void wait_async0() {
  asm volatile("s_wait_asynccnt 0x0" ::: "memory");
}

// ---------------------------------------------------------------------------
__global__ void f32_to_bf16_kernel(const float* __restrict__ in,
                                   __bf16* __restrict__ out, int n) {
  int i = blockIdx.x * blockDim.x + threadIdx.x;
  int stride = gridDim.x * blockDim.x;
  for (; i < n; i += stride) out[i] = (__bf16)in[i];
}

// ---------------------------------------------------------------------------
// NT GEMM: C[M,N] = A[M,K] * Bt[N,K]^T + bias.  Block 128x128, BK=32,
// 256 threads = 8 waves, each wave 64x32 (8 WMMA tiles).
// Global->LDS via async DMA, double-buffered so DMA overlaps WMMA.
// ---------------------------------------------------------------------------
__global__ __launch_bounds__(256) void gemm_nt_bf16(
    const __bf16* __restrict__ A,   // [M,K] row-major bf16
    const __bf16* __restrict__ Bt,  // [N,K] row-major bf16
    const float*  __restrict__ bias,// [N]
    __bf16* __restrict__ Cb,        // bf16 output (or null)
    float*  __restrict__ Cf,        // f32  output (or null)
    int M, int N, int K) {
  __shared__ __bf16 Asm[2][128 * LDG];
  __shared__ __bf16 Bsm[2][128 * LDG];
  __shared__ float  biasS[128];

  const int tid  = threadIdx.x;
  const int lane = tid & 31;
  const int wid  = tid >> 5;
  const int bm = blockIdx.y * 128;
  const int bn = blockIdx.x * 128;
  if (tid < 128) biasS[tid] = bias[bn + tid];

  const int wm = (wid & 1) * 64;   // wave M origin within block
  const int wn = (wid >> 1) * 32;  // wave N origin within block

  floatx8 acc[4][2];
  for (int mt = 0; mt < 4; ++mt)
    for (int nt = 0; nt < 2; ++nt)
      for (int r = 0; r < 8; ++r) acc[mt][nt][r] = 0.f;

  // Each thread DMAs 16 halves of A and 16 halves of B per K-slab.
  const int rowL = tid >> 1;        // 0..127
  const int c0   = (tid & 1) * 16;  // 0 or 16 (halves)
  const __bf16* ga = A  + (size_t)(bm + rowL) * K + c0;
  const __bf16* gb = Bt + (size_t)(bn + rowL) * K + c0;

  // prologue: DMA first K-slab into buffer 0
  {
    uint32_t la = lds_addr32(&Asm[0][rowL * LDG + c0]);
    uint32_t lb = lds_addr32(&Bsm[0][rowL * LDG + c0]);
    async_copy_b128(la,      ga);
    async_copy_b128(la + 16, ga + 8);
    async_copy_b128(lb,      gb);
    async_copy_b128(lb + 16, gb + 8);
  }

  for (int kk = 0; kk < K; kk += 32) {
    const int cur = (kk >> 5) & 1;
    wait_async0();      // this wave's DMA into buf[cur] done
    __syncthreads();    // everyone's DMA done; prev readers of buf[cur^1] done

    if (kk + 32 < K) {  // DMA next K-slab into the other buffer (overlaps WMMA)
      uint32_t la = lds_addr32(&Asm[cur ^ 1][rowL * LDG + c0]);
      uint32_t lb = lds_addr32(&Bsm[cur ^ 1][rowL * LDG + c0]);
      async_copy_b128(la,      ga + kk + 32);
      async_copy_b128(la + 16, ga + kk + 40);
      async_copy_b128(lb,      gb + kk + 32);
      async_copy_b128(lb + 16, gb + kk + 40);
    }

    bf16x16 af[4], bfr[2];
    for (int mt = 0; mt < 4; ++mt) af[mt] = frag_a(Asm[cur], LDG, wm + mt * 16, 0, lane);
    for (int nt = 0; nt < 2; ++nt) bfr[nt] = frag_b(Bsm[cur], LDG, wn + nt * 16, 0, lane);
    for (int mt = 0; mt < 4; ++mt)
      for (int nt = 0; nt < 2; ++nt)
        acc[mt][nt] = wmma_bf16(af[mt], bfr[nt], acc[mt][nt]);
  }

  const int rL = lane >> 4, cL = lane & 15;
  for (int mt = 0; mt < 4; ++mt)
    for (int nt = 0; nt < 2; ++nt) {
      int gn = bn + wn + nt * 16 + cL;
      float bvv = biasS[wn + nt * 16 + cL];
      for (int r = 0; r < 8; ++r) {
        int gm = bm + wm + mt * 16 + r + 8 * rL;
        float v = acc[mt][nt][r] + bvv;
        if (Cf) Cf[(size_t)gm * N + gn] = v;
        else    Cb[(size_t)gm * N + gn] = (__bf16)v;
      }
    }
}

// ---------------------------------------------------------------------------
// Chunkwise retention. Grid = B*H*4 (4 groups of 16 state rows per head),
// block 256 = 8 waves. Chunk C=64:
//   P = Q K^T ; A = P (.) gamma^(t-s) causal ; Y = A V + diag(g^(t+1)) Q S^T
//   S <- g^64 S + sum_s g^(63-s) v_s k_s^T
// ---------------------------------------------------------------------------
__global__ __launch_bounds__(256) void retention_kernel(
    const __bf16* __restrict__ qkv,   // [B*L, 3072] bf16
    const float*  __restrict__ gamma_raw,
    __bf16* __restrict__ ybuf) {      // [B*L, 1024] bf16
  __shared__ __bf16 Qh[64 * LDT];      // t x j
  __shared__ __bf16 Kh[64 * LDT];      // s x j   (x SCALE)
  __shared__ __bf16 Kt[64 * LDT];      // j x s   (x SCALE * g^(63-s))
  __shared__ __bf16 Vt[16 * LDT];      // i x s
  __shared__ __bf16 Amat[64 * LDT];    // t x s  (masked, decayed)
  __shared__ float  Sf[16 * 64];       // f32 master state (i x j)
  __shared__ __bf16 Sb[2][16 * LDT];   // bf16 state copies (double buffered)
  __shared__ float  gpow[65];

  const int tid = threadIdx.x, lane = tid & 31, wid = tid >> 5;
  const int bh = blockIdx.x >> 2;
  const int i0 = (blockIdx.x & 3) * 16;   // state-row / v-column group
  const int b = bh >> 4, h = bh & 15;

  const float g = 1.f / (1.f + __expf(-gamma_raw[h]));
  if (tid < 65) gpow[tid] = __powf(g, (float)tid);
  for (int i = tid; i < 16 * 64; i += 256) Sf[i] = 0.f;
  for (int i = tid; i < 16 * LDT; i += 256) {
    Sb[0][i] = (__bf16)0.f; Sb[1][i] = (__bf16)0.f;
  }
  __syncthreads();

  const size_t baserow = (size_t)b * 4096;
  int cur = 0;

  for (int c = 0; c < 64; ++c) {
    const int t0 = c * 64;
    {   // ---- load + convert chunk ----
      const int t  = tid >> 2;           // 0..63 (time within chunk)
      const int j0 = (tid & 3) << 4;     // 0,16,32,48
      const __bf16* row = qkv + (baserow + t0 + t) * 3072;
      // Q: straight bf16 copy (16 halves)
      uint4 q0 = *(const uint4*)(row + h * 64 + j0);
      uint4 q1 = *(const uint4*)(row + h * 64 + j0 + 8);
      *(uint4*)&Qh[t * LDT + j0]     = q0;
      *(uint4*)&Qh[t * LDT + j0 + 8] = q1;
      // K: scale, store row-major and transposed+decayed
      union { uint4 u[2]; __bf16 e[16]; } kv;
      kv.u[0] = *(const uint4*)(row + 1024 + h * 64 + j0);
      kv.u[1] = *(const uint4*)(row + 1024 + h * 64 + j0 + 8);
      const float w = gpow[63 - t];
      for (int u = 0; u < 16; ++u) {
        float kf = (float)kv.e[u] * SCALE_QK;
        Kh[t * LDT + j0 + u]      = (__bf16)kf;
        Kt[(j0 + u) * LDT + t]    = (__bf16)(kf * w);
      }
      // V: this block's 16 columns, transposed into Vt
      const int iv = (tid & 3) << 2;
      union { uint2 u; __bf16 e[4]; } vv;
      vv.u = *(const uint2*)(row + 2048 + h * 64 + i0 + iv);
      for (int u = 0; u < 4; ++u) Vt[(iv + u) * LDT + t] = vv.e[u];
    }
    __syncthreads();

    // ---- P = Q K^T, decay-mask -> Amat (16 tiles / 8 waves) ----
    for (int tt = wid * 2; tt < wid * 2 + 2; ++tt) {
      const int m0 = (tt >> 2) * 16;   // t tile
      const int n0 = (tt & 3) * 16;    // s tile
      floatx8 p; for (int r = 0; r < 8; ++r) p[r] = 0.f;
      for (int k2 = 0; k2 < 64; k2 += 32)
        p = wmma_bf16(frag_a(Qh, LDT, m0, k2, lane),
                      frag_b(Kh, LDT, n0, k2, lane), p);
      const int s = n0 + (lane & 15);
      for (int r = 0; r < 8; ++r) {
        const int t = m0 + r + ((lane >> 4) << 3);
        const int d = t - s;
        float v = (d >= 0) ? p[r] * gpow[d] : 0.f;
        Amat[t * LDT + s] = (__bf16)v;
      }
    }
    __syncthreads();

    if (wid < 4) {  // ---- Y = Amat*V + diag(g^(t+1)) * Q*S^T ----
      const int m0 = wid * 16;
      floatx8 aY, aI;
      for (int r = 0; r < 8; ++r) { aY[r] = 0.f; aI[r] = 0.f; }
      for (int k2 = 0; k2 < 64; k2 += 32) {
        aY = wmma_bf16(frag_a(Amat, LDT, m0, k2, lane),
                       frag_b(Vt,  LDT, 0,  k2, lane), aY);
        aI = wmma_bf16(frag_a(Qh,  LDT, m0, k2, lane),
                       frag_b(Sb[cur], LDT, 0, k2, lane), aI);
      }
      const int irel = lane & 15;
      for (int r = 0; r < 8; ++r) {
        const int t = m0 + r + ((lane >> 4) << 3);
        float yv = aY[r] + gpow[t + 1] * aI[r];
        ybuf[(baserow + t0 + t) * 1024 + h * 64 + i0 + irel] = (__bf16)yv;
      }
    } else {        // ---- S <- g^64 S + Vt * Kt^T ----
      const int j0 = (wid - 4) * 16;
      const int jc = j0 + (lane & 15);
      const float g64 = gpow[64];
      floatx8 s;
      for (int r = 0; r < 8; ++r) {
        const int i = r + ((lane >> 4) << 3);
        s[r] = g64 * Sf[i * 64 + jc];
      }
      for (int k2 = 0; k2 < 64; k2 += 32)
        s = wmma_bf16(frag_a(Vt, LDT, 0, k2, lane),
                      frag_b(Kt, LDT, j0, k2, lane), s);
      for (int r = 0; r < 8; ++r) {
        const int i = r + ((lane >> 4) << 3);
        Sf[i * 64 + jc] = s[r];
        Sb[cur ^ 1][i * LDT + jc] = (__bf16)s[r];
      }
    }
    __syncthreads();
    cur ^= 1;
  }
}

// ---------------------------------------------------------------------------
extern "C" void kernel_launch(void* const* d_in, const int* in_sizes, int n_in,
                              void* d_out, int out_size, void* d_ws, size_t ws_size,
                              hipStream_t stream) {
  (void)in_sizes; (void)n_in; (void)out_size; (void)ws_size;
  const float* x         = (const float*)d_in[0];  // [4,4096,1024]
  const float* Wqkv_w    = (const float*)d_in[1];  // [3072,1024]
  const float* Wqkv_b    = (const float*)d_in[2];  // [3072]
  const float* out_w     = (const float*)d_in[3];  // [1024,1024]
  const float* out_b     = (const float*)d_in[4];  // [1024]
  const float* gamma_raw = (const float*)d_in[5];  // [16]
  float* out = (float*)d_out;                      // [4,4096,1024] f32

  // Workspace layout (bf16 buffers, ~176 MB total)
  __bf16* xb    = (__bf16*)d_ws;                         // 16384*1024
  __bf16* wqkvb = xb    + (size_t)16384 * 1024;          // 3072*1024
  __bf16* outwb = wqkvb + (size_t)3072 * 1024;           // 1024*1024
  __bf16* qkvb  = outwb + (size_t)1024 * 1024;           // 16384*3072
  __bf16* ybuf  = qkvb  + (size_t)16384 * 3072;          // 16384*1024

  f32_to_bf16_kernel<<<4096, 256, 0, stream>>>(x,      xb,    16384 * 1024);
  f32_to_bf16_kernel<<<3072, 256, 0, stream>>>(Wqkv_w, wqkvb, 3072 * 1024);
  f32_to_bf16_kernel<<<1024, 256, 0, stream>>>(out_w,  outwb, 1024 * 1024);

  dim3 g1(3072 / 128, 16384 / 128);
  gemm_nt_bf16<<<g1, 256, 0, stream>>>(xb, wqkvb, Wqkv_b, qkvb, nullptr,
                                       16384, 3072, 1024);

  retention_kernel<<<4 * 16 * 4, 256, 0, stream>>>(qkvb, gamma_raw, ybuf);

  dim3 g2(1024 / 128, 16384 / 128);
  gemm_nt_bf16<<<g2, 256, 0, stream>>>(ybuf, outwb, out_b, nullptr, out,
                                       16384, 1024, 1024);
}
